// MGCC_63307817943566
// MI455X (gfx1250) — matrix operand order repaired and checked by reading, not compiled, not run on hardware
//
#include <hip/hip_runtime.h>
#include <hip/hip_bf16.h>

typedef __attribute__((ext_vector_type(16))) _Float16 v16h;
typedef __attribute__((ext_vector_type(8)))  float    v8f;

#define DEV __device__ __forceinline__

static constexpr int kB = 8;
static constexpr int kH = 96, kW = 96;
static constexpr int kN = kH * kW;     // 9216 tokens
static constexpr int kD = 256;         // channels
static constexpr int kO = 2 * kD;      // 512 output channels
static constexpr float kEps = 1e-5f;

// ---------------- wave32 block reductions ----------------
DEV float warpSum(float v){ for (int o=16;o;o>>=1) v += __shfl_xor(v,o,32); return v; }
DEV float warpMax(float v){ for (int o=16;o;o>>=1) v = fmaxf(v,__shfl_xor(v,o,32)); return v; }

DEV float blockSum(float v, float* buf){
  v = warpSum(v);
  __syncthreads();
  if ((threadIdx.x & 31) == 0) buf[threadIdx.x >> 5] = v;
  __syncthreads();
  float t = buf[0];
  const int nw = blockDim.x >> 5;
  for (int i = 1; i < nw; i++) t += buf[i];
  return t;
}
DEV float blockMax(float v, float* buf){
  v = warpMax(v);
  __syncthreads();
  if ((threadIdx.x & 31) == 0) buf[threadIdx.x >> 5] = v;
  __syncthreads();
  float t = buf[0];
  const int nw = blockDim.x >> 5;
  for (int i = 1; i < nw; i++) t = fmaxf(t, buf[i]);
  return t;
}

// ---------------- LayerNorm per token (256 ch), optional channel softmax ----------------
template<bool SMAX>
__global__ __launch_bounds__(256)
void k_ln_token(const float* __restrict__ x, const float* __restrict__ g,
                const float* __restrict__ bt, _Float16* __restrict__ nrm,
                _Float16* __restrict__ q)
{
  __shared__ float red[8];
  const long long t = blockIdx.x;
  const int c = threadIdx.x;
  float v   = x[t * kD + c];
  float mu  = blockSum(v, red) * (1.0f / kD);
  float d   = v - mu;
  float var = blockSum(d * d, red) * (1.0f / kD);
  float tn  = d * rsqrtf(var + kEps) * g[c] + bt[c];
  nrm[t * kD + c] = (_Float16)tn;
  if (SMAX) {
    float m = blockMax(tn, red);
    float e = __expf(tn - m);
    float s = blockSum(e, red);
    q[t * kD + c] = (_Float16)(e / s);
  }
}

// ---------------- f16 transpose [B,N,D] -> [B,D,N], 64x64 LDS tiles ----------------
__global__ __launch_bounds__(256)
void k_transpose(const _Float16* __restrict__ src, _Float16* __restrict__ dst)
{
  __shared__ alignas(16) _Float16 tl[64 * 72];   // pad to 72 halves/row
  const int n0 = blockIdx.x * 64;
  const int d0 = blockIdx.y * 64;
  const long long b = blockIdx.z;
  const _Float16* s = src + b * (long long)kN * kD;
  _Float16*       o = dst + b * (long long)kD * kN;
  const int tid = threadIdx.x;
  #pragma unroll
  for (int i = 0; i < 2; i++) {
    int u = tid + i * 256;
    int row = u >> 3, c8 = u & 7;
    *(uint4*)&tl[row * 72 + c8 * 8] =
      *(const uint4*)&s[(long long)(n0 + row) * kD + d0 + c8 * 8];
  }
  __syncthreads();
  union H8 { uint4 u; _Float16 h[8]; };
  #pragma unroll
  for (int i = 0; i < 2; i++) {
    int u = tid + i * 256;
    int row = u >> 3, c8 = u & 7;          // row = d offset, c8*8 = n offset
    H8 p;
    #pragma unroll
    for (int j = 0; j < 8; j++) p.h[j] = tl[(c8 * 8 + j) * 72 + row];
    *(uint4*)&o[(long long)(d0 + row) * kN + n0 + c8 * 8] = p.u;
  }
}

// ---------------- softmax over N (9216) per [b,d] row, in place on f16 ----------------
__global__ __launch_bounds__(256)
void k_colsoftmax(_Float16* __restrict__ buf)
{
  __shared__ alignas(16) _Float16 s[kN];   // 18KB
  __shared__ float red[8];
  _Float16* base = buf + (long long)blockIdx.x * kN;
  const int tid = threadIdx.x;
  for (int u = tid; u < kN / 8; u += 256)
    *(uint4*)&s[u * 8] = *(const uint4*)&base[u * 8];
  __syncthreads();
  float m = -3.0e38f;
  for (int i = tid; i < kN; i += 256) m = fmaxf(m, (float)s[i]);
  m = blockMax(m, red);
  float ss = 0.f;
  for (int i = tid; i < kN; i += 256) ss += __expf((float)s[i] - m);
  ss = blockSum(ss, red);
  const float inv = 1.0f / ss;
  for (int i = tid; i < kN; i += 256)
    base[i] = (_Float16)(__expf((float)s[i] - m) * inv);
}

// ---------------- fused 4x top-k masked softmax blend ----------------
// ctx f32 [B*D,256]; out attnT f16 [B,256(e),256(d)]  (transposed for GEMM B operand)
__global__ __launch_bounds__(256)
void k_topk(const float* __restrict__ ctx, const float* __restrict__ aw,
            _Float16* __restrict__ attnT)
{
  __shared__ float v[kD];
  __shared__ float red[8];
  const int bd = blockIdx.x;
  const int b = bd >> 8, drow = bd & 255;
  const float* row = ctx + (long long)bd * kD;
  const int e = threadIdx.x;
  float ve = row[e];
  v[e] = ve;
  __syncthreads();
  float m  = blockMax(ve, red);
  float ex = __expf(ve - m);
  int r = 0;                                   // exact jnp.top_k rank (ties -> lower index)
  for (int j = 0; j < kD; j++) {
    float vj = v[j];
    r += (vj > ve) || ((vj == ve) && (j < e));
  }
  const int ks[4] = { kD / 2, (2 * kD) / 3, (3 * kD) / 4, (4 * kD) / 5 };
  float scale = 0.f;
  #pragma unroll
  for (int i = 0; i < 4; i++) {
    float S = blockSum((r < ks[i]) ? ex : 0.f, red);
    if (r < ks[i]) scale += aw[i] / S;
  }
  attnT[(long long)b * kD * kD + (long long)e * kD + drow] = (_Float16)(ex * scale);
}

// ---------------- WMMA f16 GEMM: C[M,N] = A[M,K] x Bt[N,K]^T ----------------
// Block tile 128x128, BK=32, 256 thr = 8 waves (4x2), wave tile 32x64 -> 8 wmma/k-step.
// Tiles are fed with GLOBAL_LOAD_ASYNC_TO_LDS_B128 (ASYNCcnt) into double-buffered LDS:
// one s_barrier per k-step; next tile's async loads overlap current tile's WMMAs.
template<bool ATOMIC, bool OUTF16>
__global__ __launch_bounds__(256)
void k_gemm(const _Float16* __restrict__ A, int lda, long long sA,
            const _Float16* __restrict__ Bt, int ldb, long long sB,
            void* __restrict__ Cv, int ldc, long long sC,
            int K, int splitk)
{
  __shared__ alignas(16) _Float16 As[2][128 * 40];  // rows padded to 40 halves
  __shared__ alignas(16) _Float16 Bs[2][128 * 40];

  const int z   = blockIdx.z;
  const int b   = z / splitk;
  const int sk  = z - b * splitk;
  const int kLen  = K / splitk;
  const int k0beg = sk * kLen;
  const int k0end = k0beg + kLen;

  const _Float16* Abase = A  + (long long)b * sA + (long long)blockIdx.x * 128 * lda;
  const _Float16* Bbase = Bt + (long long)b * sB + (long long)blockIdx.y * 128 * ldb;

  const int tid  = threadIdx.x;
  const int lane = tid & 31;
  const int wid  = tid >> 5;
  const int wm0  = (wid >> 1) * 32;
  const int wn0  = (wid & 1) * 64;
  const int hi   = lane >> 4;
  const int lr   = lane & 15;
  const int row0 = tid >> 2;      // this thread copies rows {row0, row0+64}
  const int seg  = tid & 3;       // 8-half segment within the 32-wide k-slab

  // Issue the 128x32-half A and B tiles for k0 into LDS buffer p (async, no VGPR staging).
  // GVS mode: saddr = uniform tile base (SGPR pair), vaddr = per-lane byte offset.
  auto issueTile = [&](int k0, int p) {
    #pragma unroll
    for (int i = 0; i < 2; i++) {
      const int row = row0 + i * 64;
      unsigned ldsA = (unsigned)(uintptr_t)&As[p][row * 40 + seg * 8];
      unsigned offA = (unsigned)(((long long)row * lda + k0 + seg * 8) * 2);
      asm volatile("global_load_async_to_lds_b128 %0, %1, %2"
                   :: "v"(ldsA), "v"(offA), "s"(Abase) : "memory");
      unsigned ldsB = (unsigned)(uintptr_t)&Bs[p][row * 40 + seg * 8];
      unsigned offB = (unsigned)(((long long)row * ldb + k0 + seg * 8) * 2);
      asm volatile("global_load_async_to_lds_b128 %0, %1, %2"
                   :: "v"(ldsB), "v"(offB), "s"(Bbase) : "memory");
    }
  };

  v8f acc[2][4] = {};

  issueTile(k0beg, 0);
  int p = 0;
  for (int k0 = k0beg; k0 < k0end; k0 += 32) {
    asm volatile("s_wait_asynccnt 0x0" ::: "memory");  // my loads for buffer p done
    __syncthreads();                                   // everyone's buffer p visible
    if (k0 + 32 < k0end) issueTile(k0 + 32, p ^ 1);    // prefetch next tile (overlaps WMMA)

    union V16 { uint4 u[2]; v16h h; };
    v16h af[2], bf[4];
    #pragma unroll
    for (int r = 0; r < 2; r++) {              // A frag: halves0-7=K hi*8.., 8-15=+16
      int row = wm0 + r * 16 + lr;
      V16 t;
      t.u[0] = *(const uint4*)&As[p][row * 40 + hi * 8];
      t.u[1] = *(const uint4*)&As[p][row * 40 + hi * 8 + 16];
      af[r] = t.h;
    }
    #pragma unroll
    for (int c = 0; c < 4; c++) {              // B frag: halves0-7=K hi*8.., 8-15=+16
      int row = wn0 + c * 16 + lr;
      V16 t;
      t.u[0] = *(const uint4*)&Bs[p][row * 40 + hi * 8];
      t.u[1] = *(const uint4*)&Bs[p][row * 40 + hi * 8 + 16];
      bf[c] = t.h;
    }
    #pragma unroll
    for (int r = 0; r < 2; r++)
      #pragma unroll
      for (int c = 0; c < 4; c++)
        acc[r][c] = __builtin_amdgcn_wmma_f32_16x16x32_f16(
            false, af[r], false, bf[c], (short)0, acc[r][c], false, false);
    p ^= 1;
  }

  const int bm0 = blockIdx.x * 128;
  const int bn0 = blockIdx.y * 128;
  float*    Cf = (float*)Cv    + (long long)b * sC;
  _Float16* Ch = (_Float16*)Cv + (long long)b * sC;
  #pragma unroll
  for (int r = 0; r < 2; r++)
    #pragma unroll
    for (int c = 0; c < 4; c++)
      #pragma unroll
      for (int j = 0; j < 8; j++) {            // D frag: lane L -> M=8*hi+j, N=lr
        int row = bm0 + wm0 + r * 16 + hi * 8 + j;
        int col = bn0 + wn0 + c * 16 + lr;
        float vv = acc[r][c][j];
        if (ATOMIC)      atomicAdd(&Cf[(long long)row * ldc + col], vv);
        else if (OUTF16) Ch[(long long)row * ldc + col] = (_Float16)vv;
        else             Cf[(long long)row * ldc + col] = vv;
      }
}

// ---------------- final LayerNorm over 512 channels, in place on d_out ----------------
__global__ __launch_bounds__(256)
void k_out_ln(float* __restrict__ out, const float* __restrict__ rb,
              const float* __restrict__ g, const float* __restrict__ bt)
{
  const int t = blockIdx.x * 256 + threadIdx.x;   // token id
  const int b = t / kN, n = t - b * kN;
  float* base = out + (long long)b * kO * kN + n;
  float s = 0.f, s2 = 0.f;
  for (int o = 0; o < kO; o++) {
    float v = base[(long long)o * kN] + rb[o];
    s += v; s2 += v * v;
  }
  const float mu  = s  * (1.0f / kO);
  const float var = s2 * (1.0f / kO) - mu * mu;
  const float rs  = rsqrtf(var + kEps);
  for (int o = 0; o < kO; o++) {
    float v = base[(long long)o * kN] + rb[o];
    base[(long long)o * kN] = (v - mu) * rs * g[o] + bt[o];
  }
}

// ---------------- small utils ----------------
__global__ void k_zero_f32(float* p, int n){ int i = blockIdx.x*256 + threadIdx.x; if (i < n) p[i] = 0.f; }
__global__ void k_f32_to_f16(const float* a, _Float16* o, int n){ int i = blockIdx.x*256 + threadIdx.x; if (i < n) o[i] = (_Float16)a[i]; }

extern "C" void kernel_launch(void* const* d_in, const int* in_sizes, int n_in,
                              void* d_out, int out_size, void* d_ws, size_t ws_size,
                              hipStream_t stream)
{
  (void)in_sizes; (void)n_in; (void)out_size; (void)ws_size;
  const float* x1  = (const float*)d_in[0];
  const float* x2  = (const float*)d_in[1];
  const float* n1w = (const float*)d_in[2];
  const float* n1b = (const float*)d_in[3];
  const float* rw  = (const float*)d_in[4];
  const float* rb  = (const float*)d_in[5];
  const float* n2w = (const float*)d_in[6];
  const float* n2b = (const float*)d_in[7];
  const float* aw  = (const float*)d_in[8];
  float* out = (float*)d_out;

  const long long BDN = (long long)kB * kD * kN;
  char* w = (char*)d_ws;
  _Float16* q     = (_Float16*)w;  w += 2 * BDN;                 // [B,N,D] channel softmax
  _Float16* n1t   = (_Float16*)w;  w += 2 * BDN;                 // [B,D,N] values
  _Float16* keyv  = (_Float16*)w;  w += 2 * BDN;                 // [B,D,N] n2t -> N-softmax
  _Float16* stg   = (_Float16*)w;  w += 2 * BDN;                 // [B,N,D] staging
  float*    ctx   = (float*)w;     w += (long long)kB * kD * kD * 4;
  _Float16* attnT = (_Float16*)w;  w += (long long)kB * kD * kD * 2;
  _Float16* Mm    = (_Float16*)w;  w += (long long)kB * kO * kD * 2;
  _Float16* rwh   = (_Float16*)w;  w += (long long)kO * kD * 2;

  k_f32_to_f16<<<(kO * kD + 255) / 256, 256, 0, stream>>>(rw, rwh, kO * kD);

  // LN(x1) -> stg ; transpose -> n1t (values, [B,D,N])
  k_ln_token<false><<<kB * kN, 256, 0, stream>>>(x1, n1w, n1b, stg, nullptr);
  k_transpose<<<dim3(kN / 64, kD / 64, kB), 256, 0, stream>>>(stg, n1t);

  // LN(x2) -> stg + q ; transpose -> keyv ; softmax over N in place
  k_ln_token<true><<<kB * kN, 256, 0, stream>>>(x2, n1w, n1b, stg, q);
  k_transpose<<<dim3(kN / 64, kD / 64, kB), 256, 0, stream>>>(stg, keyv);
  k_colsoftmax<<<kB * kD, 256, 0, stream>>>(keyv);

  // context[b] = keyv[b] (256x9216) x n1t[b]^T  -- split-K=8 with f32 atomics
  k_zero_f32<<<(kB * kD * kD + 255) / 256, 256, 0, stream>>>(ctx, kB * kD * kD);
  k_gemm<true, false><<<dim3(kD / 128, kD / 128, kB * 8), 256, 0, stream>>>(
      keyv, kN, (long long)kD * kN, n1t, kN, (long long)kD * kN,
      ctx, kD, (long long)kD * kD, kN, 8);

  // fused 4x top-k masked softmax blend -> attnT (transposed f16)
  k_topk<<<kB * kD, 256, 0, stream>>>(ctx, aw, attnT);

  // M[b] = reproj_w (512x256) x attnSum[b]  (A shared across batch: stride 0)
  k_gemm<false, true><<<dim3(kO / 128, kD / 128, kB), 256, 0, stream>>>(
      rwh, kD, 0LL, attnT, kD, (long long)kD * kD,
      Mm, kD, (long long)kO * kD, kD, 1);

  // rp[b] = M[b] (512x256) x query[b] (256x9216) -> d_out [B,512,N] f32
  k_gemm<false, false><<<dim3(kO / 128, kN / 128, kB), 256, 0, stream>>>(
      Mm, kD, (long long)kO * kD, q, kD, (long long)kD * kN,
      out, kN, (long long)kO * kN, kD, 1);

  // final LayerNorm over 512 channels (+reproj bias), in place
  k_out_ln<<<(kB * kN) / 256, 256, 0, stream>>>(out, rb, n2w, n2b);
}
